// GATConv_19619410608391
// MI455X (gfx1250) — matrix operand order repaired and checked by reading, not compiled, not run on hardware
//
#include <hip/hip_runtime.h>
#include <cstdint>

#define IN_DIM 128
#define HC     128   // H*C
#define CD     64    // output dim (per head / final)
#define NEG_SLOPE 0.2f

typedef __attribute__((ext_vector_type(16))) __bf16 v16bf;
typedef __attribute__((ext_vector_type(8)))  __bf16 v8bf;
typedef __attribute__((ext_vector_type(8)))  float  v8f;

#define TILE_ROWS 64
#define LDX 136   // padded LDS row stride (elements) -> 272B, 16B aligned, conflict-spread
#define LDW 136

// ---------------------------------------------------------------------------
// Kernel 1: h = x @ W  via v_wmma_f32_16x16x32_bf16
// block = 128 threads (4 waves), computes 64 rows x 128 cols of h.
// ---------------------------------------------------------------------------
__global__ __launch_bounds__(128)
void gat_gemm_kernel(const float* __restrict__ x, const float* __restrict__ W,
                     float* __restrict__ h, int n_nodes) {
  __shared__ __align__(16) __bf16 xs[TILE_ROWS * LDX];
  __shared__ __align__(16) __bf16 wt[IN_DIM * LDW];

  const int tid  = threadIdx.x;
  const int row0 = blockIdx.x * TILE_ROWS;

  // Stage x tile (64x128) as bf16.
  for (int e = tid; e < TILE_ROWS * IN_DIM; e += 128) {
    int r = e >> 7, c = e & 127;
    int gr = row0 + r;
    float v = (gr < n_nodes) ? x[(size_t)gr * IN_DIM + c] : 0.f;
    xs[r * LDX + c] = (__bf16)v;
  }
  // Stage W transposed: wt[n][k] = W[k][n]  (so B fragments are K-contiguous).
  for (int e = tid; e < IN_DIM * HC; e += 128) {
    int k = e >> 7, n = e & 127;
    wt[n * LDW + k] = (__bf16)W[e];
  }
  __syncthreads();

  const int wave = tid >> 5;
  const int lane = tid & 31;
  const int l16  = lane & 15;
  const int hi   = lane >> 4;
  const int m0   = wave * 16;           // 16 rows per wave

  v8f acc[8];
#pragma unroll
  for (int i = 0; i < 8; ++i) acc[i] = v8f{};

#pragma unroll
  for (int kk = 0; kk < 4; ++kk) {      // K = 128 in 4 steps of 32
    // A fragment (16x32 bf16): lane<16 holds M=l16, K = kk*32 + {0..7,16..23};
    // lanes 16..31 hold K = kk*32 + {8..15,24..31}  (ISA 7.12.2 layout).
    union { v16bf v; v8bf p[2]; } ua;
    const __bf16* arow = &xs[(m0 + l16) * LDX + kk * 32 + hi * 8];
    ua.p[0] = *(const v8bf*)(arow);        // K offsets +0..7
    ua.p[1] = *(const v8bf*)(arow + 16);   // K offsets +16..23
#pragma unroll
    for (int nt = 0; nt < 8; ++nt) {
      // B fragment (32x16 bf16): lane holds N = nt*16+l16,
      // K = kk*32 + hi*8 + {0..7,16..23} (same order as A).
      union { v16bf v; v8bf p[2]; } ub;
      const __bf16* brow = &wt[(nt * 16 + l16) * LDW + kk * 32 + hi * 8];
      ub.p[0] = *(const v8bf*)(brow);
      ub.p[1] = *(const v8bf*)(brow + 16);
      acc[nt] = __builtin_amdgcn_wmma_f32_16x16x32_bf16(
          false, ua.v, false, ub.v, (short)0, acc[nt], false, false);
    }
  }

  // C/D layout: VGPR r, lanes 0-15 -> M=r, lanes 16-31 -> M=8+r; N = l16.
#pragma unroll
  for (int nt = 0; nt < 8; ++nt) {
#pragma unroll
    for (int r = 0; r < 8; ++r) {
      int gr = row0 + m0 + hi * 8 + r;
      if (gr < n_nodes) h[(size_t)gr * HC + nt * 16 + l16] = acc[nt][r];
    }
  }
}

// ---------------------------------------------------------------------------
// Kernel 2: a_src[n][hd] = sum_c h[n][hd][c]*att_src[hd][c]; same for dst.
// One wave32 per node; lanes 0-15 = head 0, lanes 16-31 = head 1.
// ---------------------------------------------------------------------------
__global__ __launch_bounds__(256)
void gat_attn_kernel(const float* __restrict__ h, const float* __restrict__ att_src,
                     const float* __restrict__ att_dst,
                     float* __restrict__ a_src, float* __restrict__ a_dst, int n_nodes) {
  int wave = threadIdx.x >> 5;
  int lane = threadIdx.x & 31;
  int node = blockIdx.x * 8 + wave;
  if (node >= n_nodes) return;

  const float4 hv = *(const float4*)&h[(size_t)node * HC + lane * 4];
  const float4 as = *(const float4*)&att_src[lane * 4];
  const float4 ad = *(const float4*)&att_dst[lane * 4];
  float ps = hv.x * as.x + hv.y * as.y + hv.z * as.z + hv.w * as.w;
  float pd = hv.x * ad.x + hv.y * ad.y + hv.z * ad.z + hv.w * ad.w;
#pragma unroll
  for (int o = 1; o < 16; o <<= 1) {   // reduce within 16-lane head groups
    ps += __shfl_xor(ps, o, 32);
    pd += __shfl_xor(pd, o, 32);
  }
  if ((lane & 15) == 0) {
    int hd = lane >> 4;
    a_src[node * 2 + hd] = ps;
    a_dst[node * 2 + hd] = pd;
  }
}

// ---------------------------------------------------------------------------
// Kernel 3: init amax=-inf, denom=0, out = bias (bias folded in here).
// ---------------------------------------------------------------------------
__global__ void gat_init_kernel(float* __restrict__ amax, float* __restrict__ denom,
                                float* __restrict__ out, const float* __restrict__ bias,
                                int n_nodes) {
  int i = blockIdx.x * blockDim.x + threadIdx.x;
  if (i < n_nodes * 2) { amax[i] = -__builtin_inff(); denom[i] = 0.f; }
  if (i < n_nodes * CD) out[i] = bias[i & 63];
}

__device__ __forceinline__ void atomicMaxF(float* addr, float v) {
  if (v >= 0.f) atomicMax((int*)addr, __float_as_int(v));
  else          atomicMin((unsigned int*)addr, __float_as_uint(v));
}

__device__ __forceinline__ void edge_sd(const int* __restrict__ ei, int e, int nE,
                                        int& s, int& d) {
  if (e < nE) { s = ei[e]; d = ei[nE + e]; }
  else        { s = d = e - nE; }        // self loops appended
}

// ---------------------------------------------------------------------------
// Kernel 4: alpha = leaky_relu(a_src[s]+a_dst[d]); segment max via float atomics.
// ---------------------------------------------------------------------------
__global__ void gat_edge_max_kernel(const int* __restrict__ ei,
                                    const float* __restrict__ a_src,
                                    const float* __restrict__ a_dst,
                                    float* __restrict__ alpha, float* __restrict__ amax,
                                    int nE, int n_nodes) {
  int e = blockIdx.x * blockDim.x + threadIdx.x;
  if (e >= nE + n_nodes) return;
  int s, d; edge_sd(ei, e, nE, s, d);
  float l0 = a_src[s * 2 + 0] + a_dst[d * 2 + 0];
  float l1 = a_src[s * 2 + 1] + a_dst[d * 2 + 1];
  l0 = l0 > 0.f ? l0 : NEG_SLOPE * l0;
  l1 = l1 > 0.f ? l1 : NEG_SLOPE * l1;
  alpha[e * 2 + 0] = l0;
  alpha[e * 2 + 1] = l1;
  atomicMaxF(&amax[d * 2 + 0], l0);
  atomicMaxF(&amax[d * 2 + 1], l1);
}

// ---------------------------------------------------------------------------
// Kernel 5: ex = exp(alpha - amax[dst]) (in place); denom += ex.
// Every node has a self loop, so amax is always finite (ref's isfinite fixup is a no-op).
// ---------------------------------------------------------------------------
__global__ void gat_edge_exp_kernel(const int* __restrict__ ei,
                                    float* __restrict__ alpha,
                                    const float* __restrict__ amax,
                                    float* __restrict__ denom, int nE, int n_nodes) {
  int e = blockIdx.x * blockDim.x + threadIdx.x;
  if (e >= nE + n_nodes) return;
  int s, d; edge_sd(ei, e, nE, s, d);
  float e0 = expf(alpha[e * 2 + 0] - amax[d * 2 + 0]);
  float e1 = expf(alpha[e * 2 + 1] - amax[d * 2 + 1]);
  alpha[e * 2 + 0] = e0;
  alpha[e * 2 + 1] = e1;
  atomicAdd(&denom[d * 2 + 0], e0);
  atomicAdd(&denom[d * 2 + 1], e1);
}

// ---------------------------------------------------------------------------
// Kernel 6: scatter. 64 threads per edge; head-mean folded in (0.5f factor).
// Coalesced 512B gather of h[src]; one f32 atomic per output element.
// ---------------------------------------------------------------------------
__global__ __launch_bounds__(256)
void gat_scatter_kernel(const int* __restrict__ ei, const float* __restrict__ h,
                        const float* __restrict__ ex, const float* __restrict__ denom,
                        float* __restrict__ out, int nE, int n_nodes) {
  int t = blockIdx.x * blockDim.x + threadIdx.x;
  int e = t >> 6, c = t & 63;
  if (e >= nE + n_nodes) return;
  int s, d; edge_sd(ei, e, nE, s, d);
  float c0 = ex[e * 2 + 0] / (denom[d * 2 + 0] + 1e-16f);
  float c1 = ex[e * 2 + 1] / (denom[d * 2 + 1] + 1e-16f);
  float m = 0.5f * (h[(size_t)s * HC + c] * c0 + h[(size_t)s * HC + 64 + c] * c1);
  atomicAdd(&out[(size_t)d * CD + c], m);
}

// ---------------------------------------------------------------------------
extern "C" void kernel_launch(void* const* d_in, const int* in_sizes, int n_in,
                              void* d_out, int out_size, void* d_ws, size_t ws_size,
                              hipStream_t stream) {
  const float* x       = (const float*)d_in[0];
  const float* W       = (const float*)d_in[1];
  const float* att_src = (const float*)d_in[2];
  const float* att_dst = (const float*)d_in[3];
  const float* bias    = (const float*)d_in[4];
  const int*   ei      = (const int*)d_in[5];

  const int N = in_sizes[0] / IN_DIM;
  const int E = in_sizes[5] / 2;
  const int etot = E + N;

  float* out = (float*)d_out;

  // workspace partition (floats): h | a_src | a_dst | amax | denom | ex
  float* ws    = (float*)d_ws;
  float* h     = ws;
  float* a_src = h + (size_t)N * HC;
  float* a_dst = a_src + (size_t)N * 2;
  float* amax  = a_dst + (size_t)N * 2;
  float* denom = amax + (size_t)N * 2;
  float* exbuf = denom + (size_t)N * 2;   // etot*2 floats

  gat_gemm_kernel<<<(N + TILE_ROWS - 1) / TILE_ROWS, 128, 0, stream>>>(x, W, h, N);
  gat_attn_kernel<<<(N + 7) / 8, 256, 0, stream>>>(h, att_src, att_dst, a_src, a_dst, N);
  gat_init_kernel<<<((size_t)N * CD + 255) / 256, 256, 0, stream>>>(amax, denom, out, bias, N);
  gat_edge_max_kernel<<<(etot + 255) / 256, 256, 0, stream>>>(ei, a_src, a_dst, exbuf, amax, E, N);
  gat_edge_exp_kernel<<<(etot + 255) / 256, 256, 0, stream>>>(ei, exbuf, amax, denom, E, N);
  gat_scatter_kernel<<<(int)(((size_t)etot * 64 + 255) / 256), 256, 0, stream>>>(
      ei, h, exbuf, denom, out, E, N);
}
